// HyperFidelityPriorBuilder_2954937500247
// MI455X (gfx1250) — matrix operand order repaired and checked by reading, not compiled, not run on hardware
//
#include <hip/hip_runtime.h>

#define NTOK 2048
#define NB   2
#define EDIM 384
#define DPP  16
#define TOPK_ 32

typedef __attribute__((ext_vector_type(16))) _Float16 v16h;
typedef __attribute__((ext_vector_type(8)))  float    v8f;
typedef __attribute__((ext_vector_type(2)))  _Float16 h2v;

union H16 { v16h v; unsigned int u[8]; _Float16 e[16]; };
union H2U { unsigned int u; h2v v; };

__device__ __forceinline__ float sp_(float x)   { return log1pf(__expf(x)); }
// exact-erf GELU (used only in tiny K1)
__device__ __forceinline__ float gelu_(float x) { return 0.5f*x*(1.0f+erff(x*0.70710678118654752f)); }
// branchless tanh-form GELU on native v_exp (hot path, K2)
__device__ __forceinline__ float gelu_fast(float x) {
    float u  = x*(0.7978845608028654f + 0.0356774081f*x*x);  // sqrt(2/pi)*(x + 0.044715 x^3)
    float e  = __expf(2.0f*u);
    float th = (e - 1.0f)/(e + 1.0f);
    return 0.5f*x*(1.0f + th);
}

// ---------------- K0: zero output ----------------
__global__ void k_zero(float* __restrict__ p, int n) {
    int i = blockIdx.x*blockDim.x + threadIdx.x;
    if (i < n) p[i] = 0.0f;
}

// ---------------- K1: intensity projection z = GELU(x@Wi1+b)@Wi2+b ----------------
__global__ __launch_bounds__(256) void k_z(
    const float* __restrict__ feats, const float* __restrict__ Wi1, const float* __restrict__ bi1,
    const float* __restrict__ Wi2,   const float* __restrict__ bi2,
    float* __restrict__ zf32, _Float16* __restrict__ zf16, float* __restrict__ rz)
{
    __shared__ float sW1[EDIM*32];
    __shared__ float sW2[32*DPP];
    __shared__ float sb1[32];
    __shared__ float sb2[DPP];
    for (int idx = threadIdx.x; idx < EDIM*32; idx += 256) sW1[idx] = Wi1[idx];
    for (int idx = threadIdx.x; idx < 32*DPP; idx += 256) sW2[idx] = Wi2[idx];
    if (threadIdx.x < 32)  sb1[threadIdx.x] = bi1[threadIdx.x];
    if (threadIdx.x < DPP) sb2[threadIdx.x] = bi2[threadIdx.x];
    __syncthreads();

    int t = blockIdx.x*256 + threadIdx.x;          // 0..4095 (B*N rows)
    const float* f = feats + (size_t)t*EDIM;
    float h[32];
    #pragma unroll
    for (int o = 0; o < 32; o++) h[o] = sb1[o];
    for (int e = 0; e < EDIM; e++) {
        float fv = f[e];
        #pragma unroll
        for (int o = 0; o < 32; o++) h[o] = fmaf(fv, sW1[e*32+o], h[o]);
    }
    #pragma unroll
    for (int o = 0; o < 32; o++) h[o] = gelu_(h[o]);
    float z[DPP];
    #pragma unroll
    for (int d = 0; d < DPP; d++) z[d] = sb2[d];
    #pragma unroll
    for (int o = 0; o < 32; o++) {
        float hv = h[o];
        #pragma unroll
        for (int d = 0; d < DPP; d++) z[d] = fmaf(hv, sW2[o*DPP+d], z[d]);
    }
    float s = 0.0f;
    #pragma unroll
    for (int d = 0; d < DPP; d++) {
        zf32[(size_t)t*DPP+d] = z[d];
        zf16[(size_t)t*DPP+d] = (_Float16)z[d];
        s += z[d]*z[d];
    }
    rz[t] = s;
}

// ---------------- K2: fused prior A = w0*Asp + w1*Aint + w2*Abnd + w3*offdiag ----------------
// One wave32 per 16x16 (i,j) tile. 1 WMMA for the gram (A_int), 16 WMMAs for boundary MLP.
__global__ __launch_bounds__(256) void k_prior(
    const _Float16* __restrict__ zf16, const float* __restrict__ rz,
    const float* __restrict__ pos,
    const float* __restrict__ Wb1, const float* __restrict__ bb1,
    const float* __restrict__ Wb2, const float* __restrict__ bb2,
    const float* __restrict__ lsig, const float* __restrict__ ltau,
    const float* __restrict__ lalp, const float* __restrict__ plog,
    float* __restrict__ A)
{
    const int lane = threadIdx.x & 31;
    const int wv   = threadIdx.x >> 5;
    const int t  = blockIdx.x*8 + wv;              // 0..32767 tiles
    const int b  = t >> 14;                        // 128*128 tiles per batch
    const int ti = (t >> 7) & 127;
    const int tj = t & 127;
    const int i0 = ti*16, j0 = tj*16;

    // softmax mixing weights + softplus scalars (uniform -> SALU float pipe)
    float l0 = plog[0], l1 = plog[1], l2 = plog[2], l3 = plog[3];
    float mx = fmaxf(fmaxf(l0,l1), fmaxf(l2,l3));
    float e0 = __expf(l0-mx), e1 = __expf(l1-mx), e2 = __expf(l2-mx), e3 = __expf(l3-mx);
    float einv = 1.0f/(e0+e1+e2+e3);
    float w0 = e0*einv, w1 = e1*einv, w2 = e2*einv, w3 = e3*einv;
    float sg = sp_(lsig[0]); float inv2sig = 1.0f/(2.0f*sg*sg + 1e-8f);
    float ta = sp_(ltau[0]); float inv2tau = 1.0f/(2.0f*ta*ta + 1e-8f);
    float alpha = sp_(lalp[0]);

    const int nlo   = (lane < 16) ? 1 : 0;
    const int lr    = lane & 15;
    const int base8 = nlo ? 0 : 8;
    const int j     = j0 + lr;

    // z rows (f16, 16 halves = 8 dwords per row)
    const unsigned int* zi_row = (const unsigned int*)(zf16 + (size_t)(b*NTOK + i0 + lr)*DPP);
    const unsigned int* zj_row = (const unsigned int*)(zf16 + (size_t)(b*NTOK + j)*DPP);
    unsigned int zih[8], zjl[8];
    #pragma unroll
    for (int e = 0; e < 8; e++) { zih[e] = zi_row[e]; zjl[e] = zj_row[e]; }

    // --- gram WMMA: A = Zi (M=i_local, K=d pad->32), B = Zj^T (K=d, N=j_local) ---
    H16 Ag, Bg;
    #pragma unroll
    for (int e = 0; e < 4; e++) {
        Ag.u[e]   = nlo ? zih[e] : 0u;             // constant indices -> single v_cndmask each
        Ag.u[4+e] = nlo ? zih[e+4] : 0u;
    }
    #pragma unroll
    for (int e = 0; e < 8; e++) { Bg.u[e] = nlo ? zjl[e] : 0u; }
    v8f cz = {0.f,0.f,0.f,0.f,0.f,0.f,0.f,0.f};
    v8f G = __builtin_amdgcn_wmma_f32_16x16x32_f16(false, Ag.v, false, Bg.v, (short)0, cz, false, false);

    // --- Wb1^T as A operand for boundary MLP: M=hidden, K=d ---
    H16 Aw;
    #pragma unroll
    for (int k = 0; k < 8; k++) {
        Aw.e[k]   = nlo ? (_Float16)Wb1[k*DPP + lr] : (_Float16)0.0f;
        Aw.e[8+k] = nlo ? (_Float16)Wb1[(8+k)*DPP + lr] : (_Float16)0.0f;
    }

    float rzj = rz[b*NTOK + j];
    float pjx = pos[j*3+0], pjy = pos[j*3+1], pjz = pos[j*3+2];

    float acc[8];
    #pragma unroll
    for (int r = 0; r < 8; r++) {
        int i = i0 + base8 + r;
        float d2z  = fmaxf(rz[b*NTOK+i] + rzj - 2.0f*G[r], 0.0f);
        float aint = __expf(-d2z*inv2tau);
        float dx = pos[i*3+0]-pjx, dy = pos[i*3+1]-pjy, dz = pos[i*3+2]-pjz;
        float asp = __expf(-(dx*dx+dy*dy+dz*dz)*inv2sig);
        acc[r] = (i != j) ? (w0*asp + w1*aint + w3) : 0.0f;   // A_st contributes w3 offdiag
    }

    float bb1v[8], wb2v[8];
    #pragma unroll
    for (int r = 0; r < 8; r++) { bb1v[r] = bb1[base8+r]; wb2v[r] = Wb2[base8+r]; }
    float bb2s = bb2[0];

    // --- boundary prior: 16 chunks, chunk p = row i0+p vs all 16 j's ---
    #pragma unroll
    for (int p = 0; p < 16; p++) {
        H16 Bb;
        #pragma unroll
        for (int e = 0; e < 8; e++) {
            // compile-time-uniform lane index -> v_readlane broadcast (SGPR), no ds_bpermute
            unsigned int zp = (unsigned int)__builtin_amdgcn_readlane((int)zih[e], p);
            H2U a, bv, d;
            a.u = zp; bv.u = zjl[e];
            d.v = a.v - bv.v;                                            // packed f16 sub
            Bb.u[e] = nlo ? (d.u & 0x7fff7fffu) : 0u;                    // |delta|, K pad zero
        }
        v8f Hh = __builtin_amdgcn_wmma_f32_16x16x32_f16(false, Aw.v, false, Bb.v, (short)0, cz, false, false);
        float part = 0.0f;
        #pragma unroll
        for (int r = 0; r < 8; r++)
            part += gelu_fast(Hh[r] + bb1v[r]) * wb2v[r];                // branchless GELU
        part += __shfl_xor(part, 16, 32);                                // sum across hidden halves
        float bscore = part + bb2s;
        float bnd = 1.0f/(1.0f + __expf(alpha*bscore));                  // 1 - sigmoid(alpha*bscore)
        float val = ((i0 + p) != j) ? (w2*bnd) : 0.0f;
        if ((p < 8) == (nlo != 0)) acc[p & 7] += val;                    // land into C-layout slot
    }

    #pragma unroll
    for (int r = 0; r < 8; r++) {
        int i = i0 + base8 + r;
        A[((size_t)(b*NTOK) + i)*NTOK + j] = acc[r];
    }
}

// ---------------- K3: symmetrize+clamp, top-32 per row, scatter + normalize ----------------
__global__ __launch_bounds__(128) void k_topk(const float* __restrict__ A, float* __restrict__ out)
{
    __shared__ float sv[4][NTOK];
    const int lane = threadIdx.x & 31;
    const int wv   = threadIdx.x >> 5;
    const int row  = blockIdx.x*4 + wv;            // 0..4095
    const int b = row >> 11;
    const int i = row & (NTOK-1);
    const size_t rbase = ((size_t)(b*NTOK) + i)*NTOK;
    const size_t cbase = (size_t)(b*NTOK)*NTOK + i;
    float* v = sv[wv];

    for (int c = 0; c < 64; c++) {
        int jj = lane + c*32;
        float a  = A[rbase + jj];
        float at = A[cbase + (size_t)jj*NTOK];
        float s  = fminf(fmaxf(0.5f*(a + at), 0.0f), 1.0f);
        v[jj] = s;
    }
    // per-lane running max over its 64 slots
    float lmax = -1.0f; int lidx = lane;
    for (int c = 0; c < 64; c++) {
        int jj = lane + c*32; float x = v[jj];
        if (x > lmax) { lmax = x; lidx = jj; }
    }
    float myv = 0.0f; int myi = lane; float rowsum = 0.0f;
    for (int r = 0; r < TOPK_; r++) {
        float gv = lmax; int gi = lidx;
        #pragma unroll
        for (int off = 16; off >= 1; off >>= 1) {
            float ov = __shfl_xor(gv, off, 32);
            int   oi = __shfl_xor(gi, off, 32);
            if (ov > gv || (ov == gv && oi < gi)) { gv = ov; gi = oi; }
        }
        if (lane == r) { myv = gv; myi = gi; }
        rowsum += gv;
        if ((gi & 31) == lane) {                   // owner removes winner, rescans
            v[gi] = -1.0f;
            lmax = -1.0f; lidx = lane;
            for (int c = 0; c < 64; c++) {
                int jj = lane + c*32; float x = v[jj];
                if (x > lmax) { lmax = x; lidx = jj; }
            }
        }
    }
    float inv = 1.0f/fmaxf(rowsum, 1e-8f);
    out[rbase + myi] = myv*inv;                    // 32 distinct scatter writes per row
}

extern "C" void kernel_launch(void* const* d_in, const int* in_sizes, int n_in,
                              void* d_out, int out_size, void* d_ws, size_t ws_size,
                              hipStream_t stream)
{
    (void)in_sizes; (void)n_in; (void)ws_size;
    const float* feats = (const float*)d_in[0];
    const float* pos   = (const float*)d_in[1];
    const float* Wi1   = (const float*)d_in[2];
    const float* bi1   = (const float*)d_in[3];
    const float* Wi2   = (const float*)d_in[4];
    const float* bi2   = (const float*)d_in[5];
    const float* Wb1   = (const float*)d_in[6];
    const float* bb1   = (const float*)d_in[7];
    const float* Wb2   = (const float*)d_in[8];
    const float* bb2   = (const float*)d_in[9];
    const float* lsig  = (const float*)d_in[10];
    const float* ltau  = (const float*)d_in[11];
    const float* lalp  = (const float*)d_in[12];
    // d_in[13] = log_gamma: unused in forward
    const float* plog  = (const float*)d_in[14];
    float* out = (float*)d_out;

    char* w = (char*)d_ws;
    float*    zf32  = (float*)w;                       // 4096*16*4   = 256 KB
    _Float16* zf16  = (_Float16*)(w + 262144);         // 4096*16*2   = 128 KB
    float*    rz    = (float*)(w + 393216);            // 4096*4      =  16 KB
    float*    Afull = (float*)(w + 409600);            // 2*2048*2048*4 = 32 MB

    hipLaunchKernelGGL(k_zero, dim3((out_size + 255)/256), dim3(256), 0, stream, out, out_size);
    hipLaunchKernelGGL(k_z,    dim3(16),   dim3(256), 0, stream, feats, Wi1, bi1, Wi2, bi2, zf32, zf16, rz);
    hipLaunchKernelGGL(k_prior,dim3(4096), dim3(256), 0, stream, zf16, rz, pos, Wb1, bb1, Wb2, bb2,
                       lsig, ltau, lalp, plog, Afull);
    hipLaunchKernelGGL(k_topk, dim3(1024), dim3(128), 0, stream, Afull, out);
}
